// DRMamba_79147657331180
// MI455X (gfx1250) — compile-verified, matrix operand order checked by reading, not run on hardware
//
#include <hip/hip_runtime.h>
#include <hip/hip_bf16.h>

// ---------------- types for WMMA ----------------
typedef __attribute__((ext_vector_type(16))) __bf16 v16bf;
typedef __attribute__((ext_vector_type(8)))  float  v8f;

#define LSEQ   4096
#define BATCH  8
#define DMODEL 96
#define DINNER 192
#define DSTATE 16
#define NPAD   48          // dt_rank(6) + 16 + 16 = 38, padded to 48
#define ROWS   (BATCH * LSEQ)   // 32768
#define NCHUNK 32
#define CLEN   (LSEQ / NCHUNK)  // 128

static __device__ __forceinline__ unsigned short f2bf(float f) {
    unsigned int u = __float_as_uint(f);
    unsigned int r = u + 0x7FFFu + ((u >> 16) & 1u);   // RNE
    return (unsigned short)(r >> 16);
}
static __device__ __forceinline__ float silu(float x) {
    return x / (1.f + __expf(-x));
}

// ---------------- 1) flip(ch) + transpose + f32->bf16 pack of input -------
// seq[b, l, c] = x[b, 95-c, l]
__global__ void k_pack_in(const float* __restrict__ x, unsigned short* __restrict__ seqb) {
    __shared__ float t[32][33];
    int b = blockIdx.z;
    int l = blockIdx.x * 32 + threadIdx.x;
    int c = blockIdx.y * 32 + threadIdx.y;
    t[threadIdx.y][threadIdx.x] = x[((size_t)b * DMODEL + (DMODEL - 1 - c)) * LSEQ + l];
    __syncthreads();
    int c2 = blockIdx.y * 32 + threadIdx.x;
    int l2 = blockIdx.x * 32 + threadIdx.y;
    seqb[((size_t)b * LSEQ + l2) * DMODEL + c2] = f2bf(t[threadIdx.x][threadIdx.y]);
}

// ---------------- 2) weight converts ----------------
__global__ void k_cvt(const float* __restrict__ s, unsigned short* __restrict__ d, int n) {
    int i = blockIdx.x * blockDim.x + threadIdx.x;
    if (i < n) d[i] = f2bf(s[i]);
}
// x_proj: (38,192) -> (48,192) zero-padded rows
__global__ void k_cvt_pad(const float* __restrict__ s, unsigned short* __restrict__ d) {
    int i = blockIdx.x * blockDim.x + threadIdx.x;
    if (i >= NPAD * DINNER) return;
    int r = i / DINNER, c = i % DINNER;
    d[i] = (r < 38) ? f2bf(s[r * DINNER + c]) : (unsigned short)0;
}

// ---------------- 3) generic bf16 WMMA GEMM: C[M,N] = A[M,K] * W[N,K]^T ----
__global__ void k_wmma_gemm(const unsigned short* __restrict__ A,
                            const unsigned short* __restrict__ W,
                            float* __restrict__ C,
                            int M, int N, int K) {
    int wave = (blockIdx.x * blockDim.x + threadIdx.x) >> 5;
    int lane = threadIdx.x & 31;
    int tilesN = N >> 4;
    int nTiles = (M >> 4) * tilesN;
    if (wave >= nTiles) return;                 // whole-wave uniform: EXEC stays all-ones
    int mt = wave / tilesN, nt = wave % tilesN;
    int m0 = mt << 4, n0 = nt << 4;
    int r  = lane & 15;
    int kb = (lane >> 4) << 3;                  // lanes 16..31 hold K+8 / K+24 halves
    const unsigned short* arow = A + (size_t)(m0 + r) * K;
    const unsigned short* wrow = W + (size_t)(n0 + r) * K;
    v8f acc = {};
    for (int k0 = 0; k0 < K; k0 += 32) {
        union { uint4 q[2]; v16bf v; } ua, ub;
        ua.q[0] = *(const uint4*)(arow + k0 + kb);
        ua.q[1] = *(const uint4*)(arow + k0 + kb + 16);
        ub.q[0] = *(const uint4*)(wrow + k0 + kb);
        ub.q[1] = *(const uint4*)(wrow + k0 + kb + 16);
        acc = __builtin_amdgcn_wmma_f32_16x16x32_bf16(
                  false, ua.v, false, ub.v, (short)0, acc, false, false);
    }
    int mbase = (lane >> 4) << 3;               // D: vgpr i -> row mbase+i, col = lane&15
    float* crow = C + (size_t)(m0 + mbase) * N + n0 + r;
#pragma unroll
    for (int i = 0; i < 8; ++i) crow[(size_t)i * N] = acc[i];
}

// ---------------- 4) causal depthwise conv(4) + SiLU ----------------
__global__ void k_conv(const float* __restrict__ xz, const float* __restrict__ cw,
                       const float* __restrict__ cb, float* __restrict__ u32,
                       unsigned short* __restrict__ ub) {
    int i = blockIdx.x * blockDim.x + threadIdx.x;
    if (i >= ROWS * DINNER) return;
    int d = i % DINNER;
    int l = (i / DINNER) % LSEQ;
    int b = i / (DINNER * LSEQ);
    size_t base = (size_t)b * LSEQ * 384 + d;
    float acc = cb[d];
#pragma unroll
    for (int j = 0; j < 4; ++j) {
        int lj = l - 3 + j;
        if (lj >= 0) acc = fmaf(cw[d * 4 + j], xz[base + (size_t)lj * 384], acc);
    }
    float s = silu(acc);
    u32[i] = s;
    ub[i] = f2bf(s);
}

// ---------------- 6) delta = softplus(dt @ dt_proj_w^T + b) (K=6: VALU) ----
__global__ void k_delta(const float* __restrict__ dbc, const float* __restrict__ dtw,
                        const float* __restrict__ dtb, float* __restrict__ delta) {
    int i = blockIdx.x * blockDim.x + threadIdx.x;
    if (i >= ROWS * DINNER) return;
    int d = i % DINNER;
    const float* row = dbc + (size_t)(i / DINNER) * NPAD;
    float a = dtb[d];
#pragma unroll
    for (int j = 0; j < 6; ++j) a = fmaf(row[j], dtw[d * 6 + j], a);
    delta[i] = (a > 20.f) ? a : log1pf(__expf(a));
}

// ---------------- 7) chunked selective scan ----------------
// slot = (b, d, chunk); one half-wave (16 lanes = 16 states) per slot.
// Summary buffers indexed [((b*DINNER+d)*NCHUNK+chunk)*DSTATE + n].

// Pass 1: per-chunk decay product P = prod(dA) and zero-init final state S.
__global__ void k_scan1(const float* __restrict__ dbc, const float* __restrict__ delta,
                        const float* __restrict__ u32, const float* __restrict__ A_log,
                        float* __restrict__ Pb, float* __restrict__ Sb) {
    int slot = blockIdx.x * 16 + ((threadIdx.x >> 5) << 1) + ((threadIdx.x >> 4) & 1);
    int n     = threadIdx.x & 15;
    int chunk = slot & (NCHUNK - 1);
    int d     = (slot >> 5) % DINNER;
    int b     = slot / (NCHUNK * DINNER);
    float An = -__expf(A_log[d * DSTATE + n]);
    float state = 0.f, P = 1.f;
    size_t t0 = (size_t)b * LSEQ + chunk * CLEN;
    const float* dbcB = dbc + t0 * NPAD + 6 + n;
    const float* dlB  = delta + t0 * DINNER + d;
    const float* uB   = u32   + t0 * DINNER + d;
    for (int t = 0; t < CLEN; ++t) {
        __builtin_prefetch(dbcB + (size_t)(t + 1) * NPAD, 0, 0);
        float Bn = dbcB[(size_t)t * NPAD];
        float dl = dlB[(size_t)t * DINNER];
        float uu = uB[(size_t)t * DINNER];
        float dA = __expf(dl * An);
        state = fmaf(dA, state, dl * Bn * uu);
        P *= dA;
    }
    size_t idx = ((size_t)(b * DINNER + d) * NCHUNK + chunk) * DSTATE + n;
    Pb[idx] = P;
    Sb[idx] = state;
}

// Pass 2: serial scan over the 32 chunk summaries -> carry-in per chunk.
__global__ void k_scan2(const float* __restrict__ Pb, const float* __restrict__ Sb,
                        float* __restrict__ Cb) {
    int i = blockIdx.x * blockDim.x + threadIdx.x;
    if (i >= BATCH * DINNER * DSTATE) return;
    int n  = i % DSTATE;
    int bd = i / DSTATE;                  // b*DINNER + d
    float carry = 0.f;
    size_t base = (size_t)bd * NCHUNK * DSTATE + n;
#pragma unroll
    for (int k = 0; k < NCHUNK; ++k) {
        size_t idx = base + (size_t)k * DSTATE;
        Cb[idx] = carry;
        carry = fmaf(Pb[idx], carry, Sb[idx]);
    }
}

// Pass 3: re-run each chunk seeded with its carry-in; reduce over states -> ys.
__global__ void k_scan3(const float* __restrict__ dbc, const float* __restrict__ delta,
                        const float* __restrict__ u32, const float* __restrict__ A_log,
                        const float* __restrict__ Cb, float* __restrict__ ys) {
    int slot = blockIdx.x * 16 + ((threadIdx.x >> 5) << 1) + ((threadIdx.x >> 4) & 1);
    int n     = threadIdx.x & 15;
    int chunk = slot & (NCHUNK - 1);
    int d     = (slot >> 5) % DINNER;
    int b     = slot / (NCHUNK * DINNER);
    float An = -__expf(A_log[d * DSTATE + n]);
    float state = Cb[((size_t)(b * DINNER + d) * NCHUNK + chunk) * DSTATE + n];
    size_t t0 = (size_t)b * LSEQ + chunk * CLEN;
    const float* dbcB = dbc + t0 * NPAD;
    const float* dlB  = delta + t0 * DINNER + d;
    const float* uB   = u32   + t0 * DINNER + d;
    float*       yB   = ys    + t0 * DINNER + d;
    for (int t = 0; t < CLEN; ++t) {
        const float* row = dbcB + (size_t)t * NPAD;
        __builtin_prefetch(row + NPAD, 0, 0);        // global_prefetch_b8 next row
        float Bn = row[6 + n];
        float Cn = row[22 + n];
        float dl = dlB[(size_t)t * DINNER];
        float uu = uB[(size_t)t * DINNER];
        float dA = __expf(dl * An);
        state = fmaf(dA, state, dl * Bn * uu);
        float p = state * Cn;
        p += __shfl_xor(p, 8, 16);
        p += __shfl_xor(p, 4, 16);
        p += __shfl_xor(p, 2, 16);
        p += __shfl_xor(p, 1, 16);
        if (n == 0) yB[(size_t)t * DINNER] = p;
    }
}

// ---------------- 8) epilogue: y = (ys + u*D) * silu(res) -> bf16 ----------
__global__ void k_packy(const float* __restrict__ ys, const float* __restrict__ u32,
                        const float* __restrict__ Dv, const float* __restrict__ xz,
                        unsigned short* __restrict__ yb) {
    int i = blockIdx.x * blockDim.x + threadIdx.x;
    if (i >= ROWS * DINNER) return;
    int d = i % DINNER;
    float res = xz[(size_t)(i / DINNER) * 384 + DINNER + d];
    float y = (ys[i] + u32[i] * Dv[d]) * silu(res);
    yb[i] = f2bf(y);
}

// ---------------- 10) transpose + flip back to (b, c, h, w) ----------------
// d_out[b, c, l] = outf[b, l, 95-c]
__global__ void k_unpack_out(const float* __restrict__ outf, float* __restrict__ out) {
    __shared__ float t[32][33];
    int b  = blockIdx.z;
    int l0 = blockIdx.x * 32;
    int c0 = blockIdx.y * 32;
    t[threadIdx.y][threadIdx.x] =
        outf[((size_t)b * LSEQ + l0 + threadIdx.y) * DMODEL + c0 + threadIdx.x];
    __syncthreads();
    int csrc = c0 + threadIdx.y;
    int l    = l0 + threadIdx.x;
    out[((size_t)b * DMODEL + (DMODEL - 1 - csrc)) * LSEQ + l] = t[threadIdx.x][threadIdx.y];
}

// ---------------- host ----------------
extern "C" void kernel_launch(void* const* d_in, const int* in_sizes, int n_in,
                              void* d_out, int out_size, void* d_ws, size_t ws_size,
                              hipStream_t stream) {
    const float* x        = (const float*)d_in[0];
    const float* in_proj  = (const float*)d_in[1];
    const float* conv_w   = (const float*)d_in[2];
    const float* conv_b   = (const float*)d_in[3];
    const float* x_proj   = (const float*)d_in[4];
    const float* dt_w     = (const float*)d_in[5];
    const float* dt_b     = (const float*)d_in[6];
    const float* A_log    = (const float*)d_in[7];
    const float* Dv       = (const float*)d_in[8];
    const float* out_proj = (const float*)d_in[9];
    float* out = (float*)d_out;

    auto al = [](size_t v) { return (v + 255) & ~(size_t)255; };
    char* ws = (char*)d_ws;
    size_t off = 0;
    unsigned short* seqb  = (unsigned short*)(ws + off); off = al(off + (size_t)ROWS * DMODEL * 2);
    unsigned short* wInb  = (unsigned short*)(ws + off); off = al(off + (size_t)384 * DMODEL * 2);
    unsigned short* wXb   = (unsigned short*)(ws + off); off = al(off + (size_t)NPAD * DINNER * 2);
    unsigned short* wOutb = (unsigned short*)(ws + off); off = al(off + (size_t)DMODEL * DINNER * 2);
    float* xz    = (float*)(ws + off); off = al(off + (size_t)ROWS * 384 * 4);
    float* u32   = (float*)(ws + off); off = al(off + (size_t)ROWS * DINNER * 4);
    unsigned short* ub = (unsigned short*)(ws + off); off = al(off + (size_t)ROWS * DINNER * 2);
    float* dbc   = (float*)(ws + off); off = al(off + (size_t)ROWS * NPAD * 4);
    float* delta = (float*)(ws + off); off = al(off + (size_t)ROWS * DINNER * 4);
    float* ysb   = (float*)(ws + off); off = al(off + (size_t)ROWS * DINNER * 4);
    unsigned short* yb = (unsigned short*)(ws + off); off = al(off + (size_t)ROWS * DINNER * 2);
    float* outf  = (float*)(ws + off); off = al(off + (size_t)ROWS * DMODEL * 4);
    const size_t nsum = (size_t)BATCH * DINNER * NCHUNK * DSTATE;   // 786432
    float* Pb = (float*)(ws + off); off = al(off + nsum * 4);
    float* Sb = (float*)(ws + off); off = al(off + nsum * 4);
    float* Cb = (float*)(ws + off); off = al(off + nsum * 4);
    (void)ws_size; (void)in_sizes; (void)n_in; (void)out_size;

    dim3 tb(32, 32);
    // 1) pack input
    k_pack_in<<<dim3(LSEQ / 32, DMODEL / 32, BATCH), tb, 0, stream>>>(x, seqb);
    // 2) weights -> bf16
    k_cvt<<<(384 * DMODEL + 255) / 256, 256, 0, stream>>>(in_proj, wInb, 384 * DMODEL);
    k_cvt_pad<<<(NPAD * DINNER + 255) / 256, 256, 0, stream>>>(x_proj, wXb);
    k_cvt<<<(DMODEL * DINNER + 255) / 256, 256, 0, stream>>>(out_proj, wOutb, DMODEL * DINNER);
    // 3) xz = seq @ in_proj^T   (M=32768, N=384, K=96)
    {
        int tiles = (ROWS / 16) * (384 / 16);
        k_wmma_gemm<<<(tiles + 7) / 8, 256, 0, stream>>>(seqb, wInb, xz, ROWS, 384, DMODEL);
    }
    // 4) depthwise conv + SiLU
    k_conv<<<(ROWS * DINNER + 255) / 256, 256, 0, stream>>>(xz, conv_w, conv_b, u32, ub);
    // 5) dbc = u @ x_proj^T     (M=32768, N=48, K=192)
    {
        int tiles = (ROWS / 16) * (NPAD / 16);
        k_wmma_gemm<<<(tiles + 7) / 8, 256, 0, stream>>>(ub, wXb, dbc, ROWS, NPAD, DINNER);
    }
    // 6) delta = softplus(dt_proj)
    k_delta<<<(ROWS * DINNER + 255) / 256, 256, 0, stream>>>(dbc, dt_w, dt_b, delta);
    // 7) chunked selective scan (3 passes)
    {
        int slots  = BATCH * DINNER * NCHUNK;        // 49152 half-wave tasks
        int blocks = slots / 16;                     // 16 slots per 256-thread block
        k_scan1<<<blocks, 256, 0, stream>>>(dbc, delta, u32, A_log, Pb, Sb);
        k_scan2<<<(BATCH * DINNER * DSTATE + 255) / 256, 256, 0, stream>>>(Pb, Sb, Cb);
        k_scan3<<<blocks, 256, 0, stream>>>(dbc, delta, u32, A_log, Cb, ysb);
    }
    // 8) epilogue -> bf16
    k_packy<<<(ROWS * DINNER + 255) / 256, 256, 0, stream>>>(ysb, u32, Dv, xz, yb);
    // 9) out = y @ out_proj^T   (M=32768, N=96, K=192)
    {
        int tiles = (ROWS / 16) * (DMODEL / 16);
        k_wmma_gemm<<<(tiles + 7) / 8, 256, 0, stream>>>(yb, wOutb, outf, ROWS, DMODEL, DINNER);
    }
    // 10) transpose + flip out
    k_unpack_out<<<dim3(LSEQ / 32, DMODEL / 32, BATCH), tb, 0, stream>>>(outf, out);
}
